// _MMD_loss__A_38646115730032
// MI455X (gfx1250) — matrix operand order repaired and checked by reading, not compiled, not run on hardware
//
#include <hip/hip_runtime.h>
#include <hip/hip_bf16.h>

typedef __attribute__((ext_vector_type(16))) __bf16 v16bf;
typedef __attribute__((ext_vector_type(8)))  float  v8f;

#define TILE      128
#define KCHUNK    128
#define LDS_PITCH 136   // 128 + 8 bf16 pad -> row stride 272B (16B aligned, spreads banks)

// CDNA5 async global->LDS path (bypasses VGPRs, tracked by ASYNCcnt)
#if defined(__has_builtin)
#  if __has_builtin(__builtin_amdgcn_global_load_async_to_lds_b128)
#    define USE_ASYNC_LDS 1
#  endif
#  if __has_builtin(__builtin_amdgcn_sched_barrier)
#    define SCHED_FENCE() __builtin_amdgcn_sched_barrier(0)
#  endif
#endif
#ifndef USE_ASYNC_LDS
#  define USE_ASYNC_LDS 0
#endif
#ifndef SCHED_FENCE
#  define SCHED_FENCE()
#endif

#if USE_ASYNC_LDS
typedef int v4i_t __attribute__((vector_size(16)));
typedef __attribute__((address_space(1))) v4i_t GV4;   // global (AS1) int4*
typedef __attribute__((address_space(3))) v4i_t LV4;   // LDS    (AS3) int4*
#endif

// round-to-nearest-even f32 -> bf16 bits (header-version independent)
__device__ __forceinline__ unsigned short f32_to_bf16_bits(float f) {
    unsigned int x = __float_as_uint(f);
    x += 0x7FFFu + ((x >> 16) & 1u);
    return (unsigned short)(x >> 16);
}
__device__ __forceinline__ float bf16_bits_to_f32(unsigned short b) {
    return __uint_as_float(((unsigned int)b) << 16);
}

// build a v16bf operand from two contiguous 16-byte chunks
__device__ __forceinline__ v16bf load2x16(const unsigned short* p0, const unsigned short* p1) {
    union { v16bf v; uint4 q[2]; } u;
    u.q[0] = *reinterpret_cast<const uint4*>(p0);
    u.q[1] = *reinterpret_cast<const uint4*>(p1);
    return u.v;
}

// load all 8 B fragments (one per 16-column group) for K-step kk
__device__ __forceinline__ void loadBfrags(const unsigned short* ldsB,
                                           int r, int hi, int kk, v16bf* out) {
    #pragma unroll
    for (int n = 0; n < 8; ++n) {
        const unsigned short* bp = &ldsB[(n * 16 + r) * LDS_PITCH + kk + hi * 16];
        out[n] = load2x16(bp, bp + 8);
    }
}

// ---------------------------------------------------------------------------
// Kernel 1: f32 -> bf16 conversion + row norms computed FROM THE ROUNDED data
// (so the Gram diagonal cancels exactly against the norms -> exp(0)=1).
// ---------------------------------------------------------------------------
__global__ void __launch_bounds__(256)
prep_kernel(const float* __restrict__ S, const float* __restrict__ T,
            unsigned short* __restrict__ Sb, unsigned short* __restrict__ Tb,
            float* __restrict__ nS, float* __restrict__ nT,
            int ns, int nt, int D) {
    int row = blockIdx.x;
    const float* src;
    unsigned short* dstb;
    float* dstn;
    if (row < ns) {
        src = S + (size_t)row * D; dstb = Sb + (size_t)row * D; dstn = nS + row;
    } else {
        int r = row - ns;
        src = T + (size_t)r * D; dstb = Tb + (size_t)r * D; dstn = nT + r;
    }
    float ss = 0.f;
    for (int k = threadIdx.x; k < D; k += blockDim.x) {
        unsigned short b = f32_to_bf16_bits(src[k]);
        dstb[k] = b;
        float v = bf16_bits_to_f32(b);   // norm of the *rounded* value
        ss += v * v;
    }
    #pragma unroll
    for (int off = 16; off > 0; off >>= 1) ss += __shfl_xor(ss, off, 32);
    __shared__ float red[8];
    int wave = threadIdx.x >> 5, lane = threadIdx.x & 31;
    if (lane == 0) red[wave] = ss;
    __syncthreads();
    if (threadIdx.x == 0) {
        float t = 0.f;
        #pragma unroll
        for (int w = 0; w < 8; ++w) t += red[w];
        *dstn = t;
    }
}

// ---------------------------------------------------------------------------
// Kernel 2: fused Gram-tile GEMM (bf16 WMMA, f32 accum) + exp + reduction.
// grid = (cols/128, rows/128, 3); block = 256 threads = 8 waves.
// ---------------------------------------------------------------------------
__global__ void __launch_bounds__(256)
mmd_tile_kernel(const unsigned short* __restrict__ Sb,
                const unsigned short* __restrict__ Tb,
                const float* __restrict__ nS, const float* __restrict__ nT,
                float* __restrict__ acc, int D, int ns, int nt) {
    const int z = blockIdx.z;
    const unsigned short* A  = (z == 1) ? Tb : Sb;
    const unsigned short* B  = (z == 0) ? Sb : Tb;
    const float*          An = (z == 1) ? nT : nS;
    const float*          Bn = (z == 0) ? nS : nT;
    const int rows = (z == 1) ? nt : ns;
    const int cols = (z == 0) ? ns : nt;

    const int rowBase = blockIdx.y * TILE;
    const int colBase = blockIdx.x * TILE;
    if (rowBase >= rows || colBase >= cols) return;   // block-uniform

    const int tid  = threadIdx.x;
    const int wave = tid >> 5;
    const int lane = tid & 31;
    const int r    = lane & 15;     // A row-in-frag / B column / D column
    const int hi   = lane >> 4;     // K-half selector (A/B); M-half selector (D)

    __shared__ unsigned short ldsB[TILE * LDS_PITCH];
    __shared__ float          red[8];

    v8f c[8] = {};

    const unsigned short* Arow = A + (size_t)(rowBase + wave * 16 + r) * D;

    for (int kc = 0; kc < D; kc += KCHUNK) {
        __syncthreads();
        // ---- stage B tile chunk: 128 rows x 128 K of bf16 (32KB) ----
        {
            int br = tid >> 1;                 // 0..127 : gram-column row
            int h  = (tid & 1) * 64;           // 0 or 64 : K half
            const unsigned short* gsrc = B + (size_t)(colBase + br) * D + kc + h;
            unsigned short*       ldst = &ldsB[br * LDS_PITCH + h];
#if USE_ASYNC_LDS
            #pragma unroll
            for (int j = 0; j < 8; ++j) {
                GV4* gp = (GV4*)(unsigned long long)(size_t)(gsrc + j * 8);
                LV4* lp = (LV4*)(ldst + j * 8);
                __builtin_amdgcn_global_load_async_to_lds_b128(gp, lp, 0, 0);
            }
            asm volatile("s_wait_asynccnt 0x0" ::: "memory");
#else
            // register-staged: 8 loads clause up, one wait, 8 stores
            uint4 tmp[8];
            const uint4* g = reinterpret_cast<const uint4*>(gsrc);
            #pragma unroll
            for (int j = 0; j < 8; ++j) tmp[j] = g[j];
            uint4* l = reinterpret_cast<uint4*>(ldst);
            #pragma unroll
            for (int j = 0; j < 8; ++j) l[j] = tmp[j];
#endif
        }
        __syncthreads();

        // ---- compute: per K-step -> load group | sched fence | 8 WMMAs.
        // The fence keeps all 8 B fragments live (distinct regs, clause of
        // ds_loads, partial dscnt waits); the region between fences lets the
        // scheduler interleave step kk+1's loads among step kk's WMMAs.
        const unsigned short* ArowKc = Arow + kc;
        #pragma unroll
        for (int kk = 0; kk < KCHUNK; kk += 32) {
            const unsigned short* ap = ArowKc + kk + hi * 16;
            v16bf a = load2x16(ap, ap + 8);
            v16bf b[8];
            loadBfrags(ldsB, r, hi, kk, b);
            SCHED_FENCE();
            #pragma unroll
            for (int n = 0; n < 8; ++n) {
                c[n] = __builtin_amdgcn_wmma_f32_16x16x32_bf16(
                           false, a, false, b[n], (short)0, c[n], false, false);
            }
        }
    }

    // Epilogue: D layout -> global row = rowBase + wave*16 + (v + hi*8),
    //                      global col = colBase + n*16 + r
    float x2v[8];
    const float* x2p = An + rowBase + wave * 16 + hi * 8;
    #pragma unroll
    for (int v = 0; v < 8; ++v) x2v[v] = x2p[v];

    float lsum = 0.f;
    #pragma unroll
    for (int n = 0; n < 8; ++n) {
        float y2j = Bn[colBase + n * 16 + r];
        #pragma unroll
        for (int v = 0; v < 8; ++v) {
            float sq = x2v[v] + y2j - 2.f * c[n][v];
            sq = fmaxf(sq, 0.f);
            lsum += __expf(-0.5f * sq);   // SIGMA = 1.0
        }
    }

    #pragma unroll
    for (int off = 16; off > 0; off >>= 1) lsum += __shfl_xor(lsum, off, 32);
    if (lane == 0) red[wave] = lsum;
    __syncthreads();
    if (tid == 0) {
        float t = 0.f;
        #pragma unroll
        for (int w = 0; w < 8; ++w) t += red[w];
        atomicAdd(&acc[z], t);
    }
}

// ---------------------------------------------------------------------------
// Kernel 3: combine the three accumulators into the scalar output.
// ---------------------------------------------------------------------------
__global__ void combine_kernel(const float* __restrict__ acc,
                               float* __restrict__ out, int ns, int nt) {
    if (threadIdx.x == 0 && blockIdx.x == 0) {
        double dns = (double)ns, dnt = (double)nt;
        double v = (double)acc[0] / (dns * dns)
                 + (double)acc[1] / (dnt * dnt)
                 - 2.0 * (double)acc[2] / (dns * dnt);
        out[0] = (float)v;
    }
}

extern "C" void kernel_launch(void* const* d_in, const int* in_sizes, int n_in,
                              void* d_out, int out_size, void* d_ws, size_t ws_size,
                              hipStream_t stream) {
    (void)n_in; (void)out_size; (void)ws_size;
    const float* S = (const float*)d_in[0];
    const float* T = (const float*)d_in[2];
    const int ns = in_sizes[1];               // label count == ns
    const int D  = in_sizes[0] / ns;          // 256
    const int nt = in_sizes[2] / D;           // 8192

    // workspace layout
    char* ws = (char*)d_ws;
    float* acc = (float*)ws;                              // 3 floats (+pad)
    float* nSrm = (float*)(ws + 256);
    float* nTrm = nSrm + ns;
    size_t off = 256 + (size_t)(ns + nt) * sizeof(float);
    off = (off + 255) & ~(size_t)255;
    unsigned short* Sb = (unsigned short*)(ws + off);
    unsigned short* Tb = Sb + (size_t)ns * D;

    (void)hipMemsetAsync(acc, 0, 4 * sizeof(float), stream);

    prep_kernel<<<ns + nt, 256, 0, stream>>>(S, T, Sb, Tb, nSrm, nTrm, ns, nt, D);

    int gmax = ((ns > nt ? ns : nt) + TILE - 1) / TILE;
    dim3 grid(gmax, gmax, 3);
    mmd_tile_kernel<<<grid, 256, 0, stream>>>(Sb, Tb, nSrm, nTrm, acc, D, ns, nt);

    combine_kernel<<<1, 1, 0, stream>>>(acc, (float*)d_out, ns, nt);
}
